// QuantLinear_12360915878531
// MI455X (gfx1250) — 8-bit matrix operand order repaired and checked by reading, compiled, not run on hardware
//
#include <hip/hip_runtime.h>
#include <math.h>

typedef __attribute__((ext_vector_type(8))) int v8i;
typedef __attribute__((ext_vector_type(4))) int v4i;
typedef __attribute__((ext_vector_type(2))) int v2i;

#define QW 127.0f
#define QA 255.0f

template <int ITERS>
__global__ __launch_bounds__(256) void wq_kernel_reg(
    const float* __restrict__ W, signed char* __restrict__ Wq,
    float* __restrict__ row_scales, int* __restrict__ wsum, int K)
{
    const int row = blockIdx.x;
    const int tid = threadIdx.x;
    const float* w = W + (size_t)row * K;

    __shared__ float sred[256];
    __shared__ int   sint[256];

    float vals[ITERS];
    float m = 0.0f;
#pragma unroll
    for (int j = 0; j < ITERS; ++j) {
        float v = __builtin_nontemporal_load(&w[tid + j * 256]);
        vals[j] = v;
        m = fmaxf(m, fabsf(v));
    }
    sred[tid] = m;
    __syncthreads();
    for (int s = 128; s > 0; s >>= 1) {
        if (tid < s) sred[tid] = fmaxf(sred[tid], sred[tid + s]);
        __syncthreads();
    }
    const float scale = sred[0] / QW;
    const float inv   = (scale > 0.0f) ? (1.0f / scale) : 0.0f;

    int lsum = 0;
#pragma unroll
    for (int j = 0; j < ITERS; ++j) {
        float q = rintf(vals[j] * inv);
        q = fminf(fmaxf(q, -QW), QW);
        Wq[(size_t)row * K + tid + j * 256] = (signed char)(int)q;
        lsum += (int)q;
    }
    sint[tid] = lsum;
    __syncthreads();
    for (int s = 128; s > 0; s >>= 1) {
        if (tid < s) sint[tid] += sint[tid + s];
        __syncthreads();
    }
    if (tid == 0) {
        row_scales[row] = scale;
        wsum[row] = sint[0];
    }
}

__global__ __launch_bounds__(256) void wq_kernel(
    const float* __restrict__ W, signed char* __restrict__ Wq,
    float* __restrict__ row_scales, int* __restrict__ wsum, int K)
{
    const int row = blockIdx.x;
    const int tid = threadIdx.x;
    const float* w = W + (size_t)row * K;

    __shared__ float sred[256];
    __shared__ int   sint[256];

    float m = 0.0f;
    for (int i = tid; i < K; i += 256) m = fmaxf(m, fabsf(w[i]));
    sred[tid] = m;
    __syncthreads();
    for (int s = 128; s > 0; s >>= 1) {
        if (tid < s) sred[tid] = fmaxf(sred[tid], sred[tid + s]);
        __syncthreads();
    }
    const float scale = sred[0] / QW;
    const float inv   = (scale > 0.0f) ? (1.0f / scale) : 0.0f;

    int lsum = 0;
    for (int i = tid; i < K; i += 256) {
        float q = rintf(w[i] * inv);
        q = fminf(fmaxf(q, -QW), QW);
        Wq[(size_t)row * K + i] = (signed char)(int)q;
        lsum += (int)q;
    }
    sint[tid] = lsum;
    __syncthreads();
    for (int s = 128; s > 0; s >>= 1) {
        if (tid < s) sint[tid] += sint[tid + s];
        __syncthreads();
    }
    if (tid == 0) {
        row_scales[row] = scale;
        wsum[row] = sint[0];
    }
}

template <int ITERS>
__global__ __launch_bounds__(256) void xq_kernel_reg(
    const float* __restrict__ X, const float* __restrict__ cmin,
    const float* __restrict__ cmax, signed char* __restrict__ Xq,
    float* __restrict__ col_scales, float* __restrict__ col_zps, int K)
{
    const int tok = blockIdx.x;
    const int tid = threadIdx.x;
    const float* x = X + (size_t)tok * K;

    __shared__ float smn[256];
    __shared__ float smx[256];

    float vals[ITERS];
    float mn =  3.4e38f;
    float mx = -3.4e38f;
#pragma unroll
    for (int j = 0; j < ITERS; ++j) {
        const int i = tid + j * 256;
        float v = __builtin_nontemporal_load(&x[i]);
        v = fminf(fmaxf(v, cmin[i]), cmax[i]);
        vals[j] = v;
        mn = fminf(mn, v);
        mx = fmaxf(mx, v);
    }
    smn[tid] = mn; smx[tid] = mx;
    __syncthreads();
    for (int s = 128; s > 0; s >>= 1) {
        if (tid < s) {
            smn[tid] = fminf(smn[tid], smn[tid + s]);
            smx[tid] = fmaxf(smx[tid], smx[tid + s]);
        }
        __syncthreads();
    }
    float scale = (smx[0] - smn[0]) / QA;
    if (!(scale > 0.0f)) scale = 1.0f;
    const float zp  = rintf(-128.0f - smn[0] / scale);
    const float inv = 1.0f / scale;

#pragma unroll
    for (int j = 0; j < ITERS; ++j) {
        float q = rintf(vals[j] * inv) + zp;
        q = fminf(fmaxf(q, -128.0f), 127.0f);
        Xq[(size_t)tok * K + tid + j * 256] = (signed char)(int)q;
    }
    if (tid == 0) {
        col_scales[tok] = scale;
        col_zps[tok]    = zp;
    }
}

__global__ __launch_bounds__(256) void xq_kernel(
    const float* __restrict__ X, const float* __restrict__ cmin,
    const float* __restrict__ cmax, signed char* __restrict__ Xq,
    float* __restrict__ col_scales, float* __restrict__ col_zps, int K)
{
    const int tok = blockIdx.x;
    const int tid = threadIdx.x;
    const float* x = X + (size_t)tok * K;

    __shared__ float smn[256];
    __shared__ float smx[256];

    float mn =  3.4e38f;
    float mx = -3.4e38f;
    for (int i = tid; i < K; i += 256) {
        float v = fminf(fmaxf(x[i], cmin[i]), cmax[i]);
        mn = fminf(mn, v);
        mx = fmaxf(mx, v);
    }
    smn[tid] = mn; smx[tid] = mx;
    __syncthreads();
    for (int s = 128; s > 0; s >>= 1) {
        if (tid < s) {
            smn[tid] = fminf(smn[tid], smn[tid + s]);
            smx[tid] = fmaxf(smx[tid], smx[tid + s]);
        }
        __syncthreads();
    }
    float scale = (smx[0] - smn[0]) / QA;
    if (!(scale > 0.0f)) scale = 1.0f;
    const float zp  = rintf(-128.0f - smn[0] / scale);
    const float inv = 1.0f / scale;

    for (int i = tid; i < K; i += 256) {
        float v = fminf(fmaxf(x[i], cmin[i]), cmax[i]);
        float q = rintf(v * inv) + zp;
        q = fminf(fmaxf(q, -128.0f), 127.0f);
        Xq[(size_t)tok * K + i] = (signed char)(int)q;
    }
    if (tid == 0) {
        col_scales[tok] = scale;
        col_zps[tok]    = zp;
    }
}

#define TM 4
#define TN 4

__global__ __launch_bounds__(256) void gemm_iu8_kernel(
    const signed char* __restrict__ Xq,
    const signed char* __restrict__ Wq,
    const float* __restrict__ col_scales,
    const float* __restrict__ col_zps,
    const float* __restrict__ row_scales,
    const int*   __restrict__ wsum,
    const float* __restrict__ bias,
    float* __restrict__ Y,
    int M, int Nout, int K)
{
    const int lane  = threadIdx.x & 31;
    const int wave  = threadIdx.x >> 5;
    const int wm    = wave & 1;
    const int wn    = wave >> 1;
    const int m0    = blockIdx.x * 128 + wm * (TM * 16);
    const int n0    = blockIdx.y * 256 + wn * (TN * 16);
    const int lrow  = lane & 15;
    const int lhalf = lane >> 4;

    v8i acc[TM][TN] = {};

    union Frag { v8i v; v2i d2[4]; v4i d4[2]; };

    for (int k0 = 0; k0 < K; k0 += 64) {
        v8i afrag[TM];
#pragma unroll
        for (int i = 0; i < TM; ++i) {
            const signed char* ap =
                Xq + (size_t)(m0 + i * 16 + lrow) * K + k0 + lhalf * 8;
            Frag f;
            f.d2[0] = *(const v2i*)(ap);
            f.d2[1] = *(const v2i*)(ap + 16);
            f.d2[2] = *(const v2i*)(ap + 32);
            f.d2[3] = *(const v2i*)(ap + 48);
            afrag[i] = f.v;
        }
        v8i bfrag[TN];
#pragma unroll
        for (int j = 0; j < TN; ++j) {
            const signed char* bp =
                Wq + (size_t)(n0 + j * 16 + lrow) * K + k0 + lhalf * 8;
            Frag f;
            f.d2[0] = *(const v2i*)(bp);
            f.d2[1] = *(const v2i*)(bp + 16);
            f.d2[2] = *(const v2i*)(bp + 32);
            f.d2[3] = *(const v2i*)(bp + 48);
            bfrag[j] = f.v;
        }
#pragma unroll
        for (int i = 0; i < TM; ++i)
#pragma unroll
            for (int j = 0; j < TN; ++j)
                acc[i][j] = __builtin_amdgcn_wmma_i32_16x16x64_iu8(
                     true, afrag[i],
                     true, bfrag[j],
                    acc[i][j],
                     false,  false);
    }

#pragma unroll
    for (int i = 0; i < TM; ++i) {
#pragma unroll
        for (int j = 0; j < TN; ++j) {
            const int n  = n0 + j * 16 + lrow;
            const float rs = row_scales[n];
            const float bb = bias[n];
            const int   ws = wsum[n];
#pragma unroll
            for (int e = 0; e < 8; ++e) {
                const int m = m0 + i * 16 + lhalf * 8 + e;
                const float cs = col_scales[m];
                const int   zp = (int)col_zps[m];
                const float val =
                    (float)(acc[i][j][e] - zp * ws) * cs * rs + bb;
                __builtin_nontemporal_store(val, &Y[(size_t)m * Nout + n]);
            }
        }
    }
}

extern "C" void kernel_launch(void* const* d_in, const int* in_sizes, int n_in,
                              void* d_out, int out_size, void* d_ws, size_t ws_size,
                              hipStream_t stream)
{
    const float* x     = (const float*)d_in[0];
    const float* W     = (const float*)d_in[1];
    const float* bias  = (const float*)d_in[2];
    const float* cmin  = (const float*)d_in[3];
    const float* cmax  = (const float*)d_in[4];
    float* Y           = (float*)d_out;

    const int K    = in_sizes[3];
    const int Dout = in_sizes[2];
    const int N    = in_sizes[0] / K;

    char* ws = (char*)d_ws;
    size_t off = 0;
    auto carve = [&](size_t bytes) {
        char* p = ws + off;
        off = (off + bytes + 255) & ~(size_t)255;
        return p;
    };
    signed char* Wq  = (signed char*)carve((size_t)Dout * K);
    signed char* Xq  = (signed char*)carve((size_t)N * K);
    float* row_sc    = (float*)carve((size_t)Dout * 4);
    int*   wsm       = (int*)carve((size_t)Dout * 4);
    float* col_sc    = (float*)carve((size_t)N * 4);
    float* col_zp    = (float*)carve((size_t)N * 4);
    (void)ws_size; (void)n_in; (void)out_size;

    if (K == 4096) {
        wq_kernel_reg<16><<<Dout, 256, 0, stream>>>(W, Wq, row_sc, wsm, K);
        xq_kernel_reg<16><<<N, 256, 0, stream>>>(x, cmin, cmax, Xq, col_sc, col_zp, K);
    } else {
        wq_kernel<<<Dout, 256, 0, stream>>>(W, Wq, row_sc, wsm, K);
        xq_kernel<<<N, 256, 0, stream>>>(x, cmin, cmax, Xq, col_sc, col_zp, K);
    }

    dim3 grid(N / 128, Dout / 256);
    gemm_iu8_kernel<<<grid, 256, 0, stream>>>(
        Xq, Wq, col_sc, col_zp, row_sc, wsm, bias, Y, N, Dout, K);
}
